// CuteInferLinearShift_82523501626098
// MI455X (gfx1250) — compile-verified, matrix operand order checked by reading, not run on hardware
//
#include <hip/hip_runtime.h>
#include <hip/hip_bf16.h>
#include <stdint.h>

typedef __attribute__((ext_vector_type(16))) __bf16 v16bf;
typedef __attribute__((ext_vector_type(8)))  float  v8f;
typedef int v4i_vs __attribute__((vector_size(16)));   // matches builtin's pointee type

#define TS 56          // LDS row stride in bf16 elements (112B: 16B-aligned, bank-spread)
#define BM 128
#define BN 128
#define BK 32

#if __has_builtin(__builtin_amdgcn_global_load_async_to_lds_b128)
#define ASYNC_B 1
#else
#define ASYNC_B 0
#endif

struct Frag32 { uint4 lo, hi; };

__device__ __forceinline__ unsigned short bf16_hi_rne(float f) {
  unsigned int u = __float_as_uint(f);
  u += 0x7FFFu + ((u >> 16) & 1u);
  return (unsigned short)(u >> 16);
}
__device__ __forceinline__ float bf16_to_f32(unsigned short h) {
  return __uint_as_float(((unsigned int)h) << 16);
}

// Round two floats to bf16 (RNE); return packed hi pair, set packed lo-residual pair.
// Packing done with v_perm_b32 (one instruction per pair).
__device__ __forceinline__ unsigned bf16_split_pair(float f0, float f1, unsigned& lopair) {
  unsigned u0 = __float_as_uint(f0), u1 = __float_as_uint(f1);
  unsigned r0 = u0 + 0x7FFFu + ((u0 >> 16) & 1u);
  unsigned r1 = u1 + 0x7FFFu + ((u1 >> 16) & 1u);
  unsigned hipair = __builtin_amdgcn_perm(r1, r0, 0x07060302u);
  float l0 = f0 - __uint_as_float(r0 & 0xFFFF0000u);
  float l1 = f1 - __uint_as_float(r1 & 0xFFFF0000u);
  unsigned v0 = __float_as_uint(l0), v1 = __float_as_uint(l1);
  unsigned s0 = v0 + 0x7FFFu + ((v0 >> 16) & 1u);
  unsigned s1 = v1 + 0x7FFFu + ((v1 >> 16) & 1u);
  lopair = __builtin_amdgcn_perm(s1, s0, 0x07060302u);
  return hipair;
}

// fp8 e4m3(fn) round-to-nearest-even quantize -> dequantize, saturating at 448.
__device__ __forceinline__ float fp8_e4m3_qd(float x) {
  float a = fabsf(x);
  a = fminf(a, 448.0f);
  float q;
  if (a < 0.015625f) {                       // below 2^-6: denormal grid, step 2^-9
    q = rintf(a * 512.0f) * (1.0f / 512.0f);
  } else {
    int e = (int)((__float_as_uint(a) >> 23) & 0xFFu) - 127;
    float step = ldexpf(1.0f, e - 3);        // 3 mantissa bits
    q = rintf(a / step) * step;              // /step exact (power of two); rintf = RNE
  }
  return copysignf(q, x);
}

// Per-chunk(32) double fp8 quant/dequant of W -> w_eff, split into bf16 hi/lo.
__global__ __launch_bounds__(256) void wprep_kernel(const float* __restrict__ W,
                                                    unsigned short* __restrict__ wh,
                                                    unsigned short* __restrict__ wl) {
  const size_t t = (size_t)blockIdx.x * 256 + threadIdx.x;
  const size_t base = t * 32;                // chunks are contiguous in row-major W
  float w[32], sh[32];
  float amax = 0.0f;
#pragma unroll
  for (int i = 0; i < 32; ++i) {
    w[i] = W[base + i];
    amax = fmaxf(amax, fabsf(w[i]));
  }
  float s1 = fmaxf(amax * (1.0f / 448.0f), 1e-12f);
  float amax2 = 0.0f;
#pragma unroll
  for (int i = 0; i < 32; ++i) {
    float d = fp8_e4m3_qd(w[i] / s1) * s1;
    sh[i] = w[i] - d;
    w[i] = d;
    amax2 = fmaxf(amax2, fabsf(sh[i]));
  }
  float s2 = fmaxf(amax2 * (1.0f / 448.0f), 1e-12f);
#pragma unroll
  for (int i = 0; i < 32; ++i) {
    float weff = w[i] + fp8_e4m3_qd(sh[i] / s2) * s2;
    unsigned short h = bf16_hi_rne(weff);
    float lo = weff - bf16_to_f32(h);
    wh[base + i] = h;
    wl[base + i] = bf16_hi_rne(lo);
  }
}

// y = x @ w_eff^T + bias via bf16x3 split WMMA (xh*wh + xh*wl + xl*wh).
__global__ __launch_bounds__(256) void gemm_bf16x3_kernel(
    const float* __restrict__ x, const unsigned short* __restrict__ wh,
    const unsigned short* __restrict__ wl, const float* __restrict__ bias,
    float* __restrict__ out, int M, int N, int K) {
  __shared__ unsigned short Ah[BM * TS];
  __shared__ unsigned short Al[BM * TS];
  __shared__ unsigned short Bh[2 * BN * TS];   // double-buffered for async pipeline
  __shared__ unsigned short Bl[2 * BN * TS];

  const int tid  = threadIdx.x;
  const int lane = tid & 31;
  const int wave = tid >> 5;
  const int wm   = wave >> 2;        // 0..1 : 64-row slab
  const int wn   = wave & 3;         // 0..3 : 32-col slab
  const int lr   = lane & 15;
  const int hi16 = (lane >> 4) & 1;
  const int koff = hi16 * 8;         // ISA 16-bit A/B layout: hi lanes hold K=8..15,24..31

  const int m0 = blockIdx.x * BM;
  const int n0 = blockIdx.y * BN;
  const int ksteps = K / BK;

  float4 aReg[4];
#if !ASYNC_B
  uint4 bRegH[2], bRegL[2];
#endif

  auto load_A_regs = [&](int ks) {
    const int kk = ks * BK;
#pragma unroll
    for (int j = 0; j < 4; ++j) {
      int s = tid + j * 256;
      int row = s >> 3;
      int col = (s & 7) * 4;
      aReg[j] = *(const float4*)&x[(size_t)(m0 + row) * K + kk + col];
    }
  };

  auto store_A_lds = [&]() {
#pragma unroll
    for (int j = 0; j < 4; ++j) {
      int s = tid + j * 256;
      int row = s >> 3;
      int col = (s & 7) * 4;
      float4 v = aReg[j];
      uint2 ph, pl;
      ph.x = bf16_split_pair(v.x, v.y, pl.x);
      ph.y = bf16_split_pair(v.z, v.w, pl.y);
      *(uint2*)&Ah[row * TS + col] = ph;
      *(uint2*)&Al[row * TS + col] = pl;
    }
  };

#if ASYNC_B
  auto async_B = [&](int ks, int buf) {
    const int kk = ks * BK;
    unsigned short* bh = Bh + buf * (BN * TS);
    unsigned short* bl = Bl + buf * (BN * TS);
#pragma unroll
    for (int j = 0; j < 2; ++j) {
      int s = tid + j * 256;
      int row = s >> 2;
      int col = (s & 3) * 8;
      const unsigned short* gh = &wh[(size_t)(n0 + row) * K + kk + col];
      const unsigned short* gl = &wl[(size_t)(n0 + row) * K + kk + col];
      unsigned short* lh = &bh[row * TS + col];
      unsigned short* ll = &bl[row * TS + col];
      // global (addrspace 1) -> LDS (addrspace 3): low 32 bits of a flat shared
      // pointer are the LDS byte offset (aperture table, CDNA5 ISA 10.2).
      __builtin_amdgcn_global_load_async_to_lds_b128(
          (__attribute__((address_space(1))) v4i_vs*)(uintptr_t)gh,
          (__attribute__((address_space(3))) v4i_vs*)(unsigned)(uintptr_t)lh, 0, 0);
      __builtin_amdgcn_global_load_async_to_lds_b128(
          (__attribute__((address_space(1))) v4i_vs*)(uintptr_t)gl,
          (__attribute__((address_space(3))) v4i_vs*)(unsigned)(uintptr_t)ll, 0, 0);
    }
  };
#else
  auto load_B_regs = [&](int ks) {
    const int kk = ks * BK;
#pragma unroll
    for (int j = 0; j < 2; ++j) {
      int s = tid + j * 256;
      int row = s >> 2;
      int col = (s & 3) * 8;
      bRegH[j] = *(const uint4*)&wh[(size_t)(n0 + row) * K + kk + col];
      bRegL[j] = *(const uint4*)&wl[(size_t)(n0 + row) * K + kk + col];
    }
  };
  auto store_B_lds = [&]() {
#pragma unroll
    for (int j = 0; j < 2; ++j) {
      int s = tid + j * 256;
      int row = s >> 2;
      int col = (s & 3) * 8;
      *(uint4*)&Bh[row * TS + col] = bRegH[j];
      *(uint4*)&Bl[row * TS + col] = bRegL[j];
    }
  };
#endif

  auto frag = [&](const unsigned short* t, int r) -> v16bf {
    Frag32 f;
    f.lo = *(const uint4*)&t[r * TS + koff];
    f.hi = *(const uint4*)&t[r * TS + koff + 16];
    return __builtin_bit_cast(v16bf, f);
  };

  v8f acc[4][2];
#pragma unroll
  for (int i = 0; i < 4; ++i)
#pragma unroll
    for (int j = 0; j < 2; ++j) acc[i][j] = (v8f){};

  load_A_regs(0);
#if ASYNC_B
  async_B(0, 0);
#else
  load_B_regs(0);
#endif

  for (int ks = 0; ks < ksteps; ++ks) {
    __syncthreads();                       // prior compute done: A LDS + other B buffer free
#if ASYNC_B
    if (ks + 1 < ksteps) async_B(ks + 1, (ks + 1) & 1);
#endif
    store_A_lds();
#if ASYNC_B
    // Async loads complete in order: <=4 outstanding => current stage's 4 are done.
    if (ks + 1 < ksteps) __builtin_amdgcn_s_wait_asynccnt(4);
    else                 __builtin_amdgcn_s_wait_asynccnt(0);
#else
    store_B_lds();
#endif
    __syncthreads();
    if (ks + 1 < ksteps) {
      load_A_regs(ks + 1);
#if !ASYNC_B
      load_B_regs(ks + 1);
#endif
    }
    if (ks + 2 < ksteps) {
      __builtin_prefetch(&x[(size_t)(m0 + (tid >> 3)) * K + (ks + 2) * BK + (tid & 7) * 4], 0, 1);
    }

#if ASYNC_B
    const unsigned short* bhT = Bh + (ks & 1) * (BN * TS);
    const unsigned short* blT = Bl + (ks & 1) * (BN * TS);
#else
    const unsigned short* bhT = Bh;
    const unsigned short* blT = Bl;
#endif

    v16bf bhf[2], blf[2];
#pragma unroll
    for (int nt = 0; nt < 2; ++nt) {
      int r = wn * 32 + nt * 16 + lr;
      bhf[nt] = frag(bhT, r);
      blf[nt] = frag(blT, r);
    }
#pragma unroll
    for (int mt = 0; mt < 4; ++mt) {
      int r = wm * 64 + mt * 16 + lr;
      v16bf ah = frag(Ah, r);
      v16bf al = frag(Al, r);
#pragma unroll
      for (int nt = 0; nt < 2; ++nt) {
        acc[mt][nt] = __builtin_amdgcn_wmma_f32_16x16x32_bf16(
            false, ah, false, bhf[nt], (short)0, acc[mt][nt], false, false);
        acc[mt][nt] = __builtin_amdgcn_wmma_f32_16x16x32_bf16(
            false, ah, false, blf[nt], (short)0, acc[mt][nt], false, false);
        acc[mt][nt] = __builtin_amdgcn_wmma_f32_16x16x32_bf16(
            false, al, false, bhf[nt], (short)0, acc[mt][nt], false, false);
      }
    }
  }

  // Epilogue: C layout = VGPR r -> M = r + 8*hi16, N = lane&15.
#pragma unroll
  for (int nt = 0; nt < 2; ++nt) {
    int n = n0 + wn * 32 + nt * 16 + lr;
    float bv = bias[n];
#pragma unroll
    for (int mt = 0; mt < 4; ++mt) {
      int mbase = m0 + wm * 64 + mt * 16 + hi16 * 8;
#pragma unroll
      for (int r = 0; r < 8; ++r) {
        out[(size_t)(mbase + r) * N + n] = acc[mt][nt][r] + bv;
      }
    }
  }
}

extern "C" void kernel_launch(void* const* d_in, const int* in_sizes, int n_in,
                              void* d_out, int out_size, void* d_ws, size_t ws_size,
                              hipStream_t stream) {
  const float* x    = (const float*)d_in[0];
  const float* W    = (const float*)d_in[1];
  const float* bias = (const float*)d_in[2];
  float* out = (float*)d_out;

  const int N = in_sizes[2];
  const int K = in_sizes[1] / N;
  const int M = in_sizes[0] / K;

  unsigned short* wh = (unsigned short*)d_ws;
  unsigned short* wl = wh + (size_t)N * K;

  const int chunks = (int)(((long long)N * K) / 32);
  wprep_kernel<<<chunks / 256, 256, 0, stream>>>(W, wh, wl);

  dim3 grid(M / BM, N / BN);
  gemm_bf16x3_kernel<<<grid, 256, 0, stream>>>(x, wh, wl, bias, out, M, N, K);
}
